// DeepGCN_45440753992390
// MI455X (gfx1250) — matrix operand order repaired and checked by reading, not compiled, not run on hardware
//
#include <hip/hip_runtime.h>
#include <hip/hip_bf16.h>

// ---------------- problem constants (from reference) ----------------
#define NN    4000
#define DD    300
#define D2X   600     // 2*D
#define EEN   8000    // E
#define ENE   12000   // E + N (with self loops)
#define LLN   28
#define FEATF 256
#define GGN   128
#define FHN   128     // FEAT/2

// padded GEMM extents (K -> mult of 32, N -> mult of 32)
#define KP1   320     // pad of K=300
#define KP2   608     // pad of K=600
#define NP1   608     // pad of N=600
#define NP2   320     // pad of N=300
#define NPF   256     // N=256 already padded

typedef __attribute__((ext_vector_type(16))) _Float16 v16h;
typedef __attribute__((ext_vector_type(8)))  _Float16 v8h;
typedef __attribute__((ext_vector_type(8)))  float    v8f;

// ---- ordered-uint encoding so unsigned atomicMax == float max ----
__device__ __forceinline__ unsigned fenc(float f) {
  unsigned u = __float_as_uint(f);
  return (u & 0x80000000u) ? ~u : (u | 0x80000000u);
}
__device__ __forceinline__ float fdec(unsigned e) {
  return (e & 0x80000000u) ? __uint_as_float(e ^ 0x80000000u)
                           : __uint_as_float(~e);
}
#define ENC_NEG_INF 0x007FFFFFu   // fenc(-inf)

// =====================================================================
// f16 pack/pad kernels (feed the WMMA GEMM with guard-free operands)
// =====================================================================
__global__ void cvt_a_k(const float* __restrict__ A, _Float16* __restrict__ Ah,
                        int M, int K, int Kp)
{
  const long i = (long)blockIdx.x * blockDim.x + threadIdx.x;
  if (i >= (long)M * Kp) return;
  const int m = (int)(i / Kp), k = (int)(i % Kp);
  Ah[i] = (k < K) ? (_Float16)A[(size_t)m * K + k] : (_Float16)0.f;
}

__global__ void cvt_w_k(const float* __restrict__ W, _Float16* __restrict__ Wh,
                        int K, int N, int Kp, int Np)
{
  const long i = (long)blockIdx.x * blockDim.x + threadIdx.x;
  if (i >= (long)Kp * Np) return;
  const int k = (int)(i / Np), n = (int)(i % Np);
  Wh[i] = (k < K && n < N) ? (_Float16)W[(size_t)k * N + n] : (_Float16)0.f;
}

// =====================================================================
// WMMA GEMM: C[M,N] = (accum ? C : 0) + A[M,Kp] @ B[Kp,Np] + bias[N]
// f16 padded operands, fp32 C. One wave owns a 16x32 C strip (two
// 16x16 WMMA tiles sharing the A fragment), 4 waves/block along M.
// Inner loop: 2x global_load_b128 (A) + 32 strided f16 loads (B) + 2x v_wmma.
// Lane layouts per CDNA5 ISA 7.12.2.
// =====================================================================
__global__ void wmma_gemm(const _Float16* __restrict__ A, int lda,  // lda = Kp
                          const _Float16* __restrict__ B, int ldb,  // ldb = Np
                          const float* __restrict__ bias,
                          float* __restrict__ C, int ldc,
                          int M, int N, int Kp, int accum)
{
  const int lane = threadIdx.x & 31;
  const int wv   = threadIdx.x >> 5;
  const int tM   = blockIdx.y * 4 + wv;   // wave-uniform
  const int tN2  = blockIdx.x;            // 32-wide N strip
  if (tM * 16 >= M) return;               // whole-wave exit: EXEC stays all-1s

  const int hs  = lane >> 4;              // half-select (0/1)
  const int l16 = lane & 15;

  // A fragment: lane holds row M=l16; K = 8*hs+{0..7} and 16+8*hs+{0..7}
  const _Float16* Arow = A + (size_t)(tM * 16 + l16) * lda + 8 * hs;
  // B fragment: lane holds N column l16; K = 8*hs+{0..7} and 16+8*hs+{0..7}
  const _Float16* Bp   = B + (size_t)(8 * hs) * ldb + tN2 * 32 + l16;

  const int ccol0 = tN2 * 32 + l16;       // tile 0 column of this lane
  const int ccol1 = ccol0 + 16;           // tile 1 column of this lane

  v8f acc0 = {0.f, 0.f, 0.f, 0.f, 0.f, 0.f, 0.f, 0.f};
  v8f acc1 = acc0;
  if (accum) {
    if (ccol0 < N) {
#pragma unroll
      for (int r = 0; r < 8; ++r)
        acc0[r] = C[(size_t)(tM * 16 + r + 8 * hs) * ldc + ccol0];
    }
    if (ccol1 < N) {
#pragma unroll
      for (int r = 0; r < 8; ++r)
        acc1[r] = C[(size_t)(tM * 16 + r + 8 * hs) * ldc + ccol1];
    }
  }

  for (int k0 = 0; k0 < Kp; k0 += 32) {
    const v8h alo = *(const v8h*)(Arow + k0);
    const v8h ahi = *(const v8h*)(Arow + k0 + 16);
    const v16h a  = __builtin_shufflevector(alo, ahi, 0, 1, 2, 3, 4, 5, 6, 7,
                                            8, 9, 10, 11, 12, 13, 14, 15);
    const _Float16* Bk = Bp + (size_t)k0 * ldb;
    v16h b0, b1;
#pragma unroll
    for (int s = 0; s < 16; ++s) {
      const size_t kk = (size_t)(s < 8 ? s : s + 8) * ldb;
      b0[s] = Bk[kk];
      b1[s] = Bk[kk + 16];
    }
    if (k0 + 32 < Kp) {                    // CDNA5 global_prefetch_b8
      __builtin_prefetch(Arow + k0 + 32, 0, 0);
      __builtin_prefetch(Bk + (size_t)32 * ldb, 0, 0);
    }
    acc0 = __builtin_amdgcn_wmma_f32_16x16x32_f16(
        false, a, false, b0, (short)0, acc0, false, false);
    acc1 = __builtin_amdgcn_wmma_f32_16x16x32_f16(
        false, a, false, b1, (short)0, acc1, false, false);
  }

  if (ccol0 < N) {
    const float bv = bias ? bias[ccol0] : 0.f;
#pragma unroll
    for (int r = 0; r < 8; ++r)
      C[(size_t)(tM * 16 + r + 8 * hs) * ldc + ccol0] = acc0[r] + bv;
  }
  if (ccol1 < N) {
    const float bv = bias ? bias[ccol1] : 0.f;
#pragma unroll
    for (int r = 0; r < 8; ++r)
      C[(size_t)(tM * 16 + r + 8 * hs) * ldc + ccol1] = acc1[r] + bv;
  }
}

// =====================================================================
// relu(layernorm(x)) over rows of width W (one block / row, 128 thr)
// =====================================================================
__global__ void ln_relu(const float* __restrict__ X, float* __restrict__ Y,
                        const float* __restrict__ g, const float* __restrict__ b,
                        int W)
{
  __shared__ float s1[128], s2[128];
  const int row = blockIdx.x;
  const float* x = X + (size_t)row * W;
  float sum = 0.f, sq = 0.f;
  for (int i = threadIdx.x; i < W; i += 128) {
    const float v = x[i];
    sum += v; sq += v * v;
  }
  s1[threadIdx.x] = sum; s2[threadIdx.x] = sq;
  __syncthreads();
  for (int off = 64; off > 0; off >>= 1) {
    if (threadIdx.x < off) {
      s1[threadIdx.x] += s1[threadIdx.x + off];
      s2[threadIdx.x] += s2[threadIdx.x + off];
    }
    __syncthreads();
  }
  const float mean = s1[0] / (float)W;
  const float var  = s2[0] / (float)W - mean * mean;
  const float inv  = rsqrtf(var + 1e-5f);
  for (int i = threadIdx.x; i < W; i += 128) {
    const float v = (x[i] - mean) * inv * g[i] + b[i];
    Y[(size_t)row * W + i] = fmaxf(v, 0.f);
  }
}

// ---------------- node embedding: h = emb1[x0] + emb2[x1] ----------------
__global__ void embed_k(const int* __restrict__ x,
                        const float* __restrict__ e1, const float* __restrict__ e2,
                        float* __restrict__ h)
{
  const long i = (long)blockIdx.x * blockDim.x + threadIdx.x;
  if (i >= (long)NN * DD) return;
  const int n = (int)(i / DD), d = (int)(i % DD);
  h[i] = e1[(size_t)x[2 * n] * DD + d] + e2[(size_t)x[2 * n + 1] * DD + d];
}

// ---------------- per-layer segment buffers init ----------------
__global__ void seg_init(unsigned* __restrict__ menc, float* __restrict__ den,
                         float* __restrict__ out, const float* __restrict__ gin)
{
  const long i = (long)blockIdx.x * blockDim.x + threadIdx.x;
  if (i >= (long)NN * DD) return;
  menc[i] = ENC_NEG_INF;
  den[i]  = 0.f;
  out[i]  = gin[i];      // residual "+ h" inside genconv
}

__device__ __forceinline__ void edge_decode(int e, const int* ei, const int* ea,
                                            int& src, int& dst, int& a0, int& a1)
{
  if (e < EEN) { src = ei[e]; dst = ei[EEN + e]; a0 = ea[2 * e]; a1 = ea[2 * e + 1]; }
  else         { src = dst = e - EEN; a0 = 4; a1 = 0; }
}

// pass 1: msg = relu(h[src]+eemb)+eps ; segment_max(msg*t)
__global__ void msg_k(const float* __restrict__ gin, const int* __restrict__ ei,
                      const int* __restrict__ ea, const float* __restrict__ ee1,
                      const float* __restrict__ ee2, const float* __restrict__ t,
                      int l, float* __restrict__ msg, unsigned* __restrict__ menc)
{
  const long i = (long)blockIdx.x * blockDim.x + threadIdx.x;
  if (i >= (long)ENE * DD) return;
  const int e = (int)(i / DD), d = (int)(i % DD);
  int src, dst, a0, a1; edge_decode(e, ei, ea, src, dst, a0, a1);
  float m = gin[(size_t)src * DD + d] + ee1[(size_t)a0 * DD + d]
          + ee2[(size_t)a1 * DD + d];
  m = fmaxf(m, 0.f) + 1e-7f;
  msg[i] = m;
  atomicMax(&menc[(size_t)dst * DD + d], fenc(m * t[l]));
}

// pass 2: den = segment_sum(exp(s - max))
__global__ void den_k(const float* __restrict__ msg, const int* __restrict__ ei,
                      const int* __restrict__ ea, const unsigned* __restrict__ menc,
                      float* __restrict__ den, const float* __restrict__ t, int l)
{
  const long i = (long)blockIdx.x * blockDim.x + threadIdx.x;
  if (i >= (long)ENE * DD) return;
  const int e = (int)(i / DD), d = (int)(i % DD);
  int src, dst, a0, a1; edge_decode(e, ei, ea, src, dst, a0, a1);
  const size_t di = (size_t)dst * DD + d;
  atomicAdd(&den[di], __expf(msg[i] * t[l] - fdec(menc[di])));
}

// pass 3: out += msg * exp(s - max) / den
__global__ void agg_k(const float* __restrict__ msg, const int* __restrict__ ei,
                      const int* __restrict__ ea, const unsigned* __restrict__ menc,
                      const float* __restrict__ den, float* __restrict__ out,
                      const float* __restrict__ t, int l)
{
  const long i = (long)blockIdx.x * blockDim.x + threadIdx.x;
  if (i >= (long)ENE * DD) return;
  const int e = (int)(i / DD), d = (int)(i % DD);
  int src, dst, a0, a1; edge_decode(e, ei, ea, src, dst, a0, a1);
  const size_t di = (size_t)dst * DD + d;
  const float ex = __expf(msg[i] * t[l] - fdec(menc[di]));
  atomicAdd(&out[di], msg[i] * ex / den[di]);
}

// ---------------- pooling + head ----------------
__global__ void zero_k(float* __restrict__ p, long n)
{
  const long i = (long)blockIdx.x * blockDim.x + threadIdx.x;
  if (i < n) p[i] = 0.f;
}

__global__ void pool_k(const float* __restrict__ hf, const int* __restrict__ batch,
                       float* __restrict__ pooled, float* __restrict__ counts)
{
  const long i = (long)blockIdx.x * blockDim.x + threadIdx.x;
  if (i >= (long)NN * FEATF) return;
  const int n = (int)(i / FEATF), f = (int)(i % FEATF);
  const int g = batch[n];
  atomicAdd(&pooled[(size_t)g * FEATF + f], hf[i]);
  if (f == 0) atomicAdd(&counts[g], 1.f);
}

__global__ void pdiv_k(float* __restrict__ pooled, const float* __restrict__ counts)
{
  const long i = (long)blockIdx.x * blockDim.x + threadIdx.x;
  if (i >= (long)GGN * FEATF) return;
  pooled[i] /= fmaxf(counts[i / FEATF], 1.f);
}

__global__ void pred1_k(const float* __restrict__ pooled, const float* __restrict__ w,
                        const float* __restrict__ b, float* __restrict__ z)
{
  const int i = blockIdx.x * blockDim.x + threadIdx.x;
  if (i >= GGN * FHN) return;
  const int g = i / FHN, j = i % FHN;
  float acc = b[j];
  for (int k = 0; k < FEATF; ++k)
    acc += pooled[(size_t)g * FEATF + k] * w[(size_t)k * FHN + j];
  z[i] = fmaxf(acc, 0.f);
}

__global__ void pred2_k(const float* __restrict__ z, const float* __restrict__ w,
                        const float* __restrict__ b, float* __restrict__ pred)
{
  const int i = blockIdx.x * blockDim.x + threadIdx.x;
  if (i >= GGN * 2) return;
  const int g = i / 2, o = i % 2;
  float acc = b[o];
  for (int k = 0; k < FHN; ++k)
    acc += z[(size_t)g * FHN + k] * w[(size_t)k * 2 + o];
  pred[i] = acc;
}

// =====================================================================
static inline int cdiv(long a, long b) { return (int)((a + b - 1) / b); }

extern "C" void kernel_launch(void* const* d_in, const int* in_sizes, int n_in,
                              void* d_out, int out_size, void* d_ws, size_t ws_size,
                              hipStream_t stream)
{
  const int*   x          = (const int*)  d_in[0];
  const int*   edge_index = (const int*)  d_in[1];
  const int*   edge_attr  = (const int*)  d_in[2];
  const int*   batch      = (const int*)  d_in[3];
  const float* x_emb1     = (const float*)d_in[4];
  const float* x_emb2     = (const float*)d_in[5];
  const float* edge_emb1  = (const float*)d_in[6];   // (L,6,D)
  const float* edge_emb2  = (const float*)d_in[7];   // (L,3,D)
  const float* t          = (const float*)d_in[8];   // (L,)
  const float* mlp_w1     = (const float*)d_in[9];   // (L,D,2D)
  const float* mlp_b1     = (const float*)d_in[10];
  const float* mlp_g      = (const float*)d_in[11];
  const float* mlp_beta   = (const float*)d_in[12];
  const float* mlp_w2     = (const float*)d_in[13];  // (L,2D,D)
  const float* mlp_b2     = (const float*)d_in[14];
  const float* blk_g      = (const float*)d_in[15];  // (L,D)
  const float* blk_b      = (const float*)d_in[16];
  const float* feat_w     = (const float*)d_in[17];  // (D,FEAT)
  const float* feat_b     = (const float*)d_in[18];
  const float* p1_w       = (const float*)d_in[19];  // (FEAT,FEAT/2)
  const float* p1_b       = (const float*)d_in[20];
  const float* p2_w       = (const float*)d_in[21];  // (FEAT/2,2)
  const float* p2_b       = (const float*)d_in[22];
  (void)in_sizes; (void)n_in; (void)out_size; (void)ws_size;

  // ---- workspace carve (≈70 MB) ----
  char* ws = (char*)d_ws;
  size_t off = 0;
  auto carve = [&](size_t bytes) -> char* {
    off = (off + 255) & ~(size_t)255;
    char* p = ws + off; off += bytes; return p;
  };
  float*    h      = (float*)    carve((size_t)NN * DD * 4);
  float*    hn     = (float*)    carve((size_t)NN * DD * 4);
  float*    outb   = (float*)    carve((size_t)NN * DD * 4);
  unsigned* menc   = (unsigned*) carve((size_t)NN * DD * 4);
  float*    den    = (float*)    carve((size_t)NN * DD * 4);
  float*    msg    = (float*)    carve((size_t)ENE * DD * 4);
  float*    tmp1   = (float*)    carve((size_t)NN * D2X * 4);
  float*    hdn    = (float*)    carve((size_t)NN * D2X * 4);
  float*    hfeat  = (float*)    carve((size_t)NN * FEATF * 4);
  float*    zbuf   = (float*)    carve((size_t)GGN * FHN * 4);
  float*    counts = (float*)    carve((size_t)GGN * 4);
  _Float16* ah1    = (_Float16*) carve((size_t)NN * KP1 * 2);   // M x 320
  _Float16* ah2    = (_Float16*) carve((size_t)NN * KP2 * 2);   // M x 608
  _Float16* w1h    = (_Float16*) carve((size_t)KP1 * NP1 * 2);  // 320 x 608
  _Float16* w2h    = (_Float16*) carve((size_t)KP2 * NP2 * 2);  // 608 x 320

  const long ndtot = (long)NN * DD;
  const long edtot = (long)ENE * DD;
  const int  bND   = cdiv(ndtot, 256);
  const int  bED   = cdiv(edtot, 256);

  // node embedding
  embed_k<<<bND, 256, 0, stream>>>(x, x_emb1, x_emb2, h);

  // GEMM grids: M=4000 -> 250 tiles -> 63 blocks of 4 waves; x = Np/32
  dim3 g1(NP1 / 32, cdiv(NN / 16, 4));   // 19 x 63
  dim3 g2(NP2 / 32, cdiv(NN / 16, 4));   // 10 x 63
  dim3 g3(NPF / 32, cdiv(NN / 16, 4));   //  8 x 63

  const int bA1 = cdiv((long)NN * KP1, 256);
  const int bA2 = cdiv((long)NN * KP2, 256);
  const int bW1 = cdiv((long)KP1 * NP1, 256);
  const int bW2 = cdiv((long)KP2 * NP2, 256);
  const int bWF = cdiv((long)KP1 * NPF, 256);

  for (int l = 0; l < LLN; ++l) {
    const float* gin = h;
    if (l > 0) {  // pre-norm branch of the scan body
      ln_relu<<<NN, 128, 0, stream>>>(h, hn, blk_g + (size_t)l * DD,
                                      blk_b + (size_t)l * DD, DD);
      gin = hn;
    }
    // softmax edge aggregation into outb (= segsum + gin)
    seg_init<<<bND, 256, 0, stream>>>(menc, den, outb, gin);
    msg_k<<<bED, 256, 0, stream>>>(gin, edge_index, edge_attr,
                                   edge_emb1 + (size_t)l * 6 * DD,
                                   edge_emb2 + (size_t)l * 3 * DD,
                                   t, l, msg, menc);
    den_k<<<bED, 256, 0, stream>>>(msg, edge_index, edge_attr, menc, den, t, l);
    agg_k<<<bED, 256, 0, stream>>>(msg, edge_index, edge_attr, menc, den, outb, t, l);

    // MLP: tmp1 = outb @ w1 + b1 ; hdn = relu(ln(tmp1)) ; h (+)= hdn @ w2 + b2
    cvt_a_k<<<bA1, 256, 0, stream>>>(outb, ah1, NN, DD, KP1);
    cvt_w_k<<<bW1, 256, 0, stream>>>(mlp_w1 + (size_t)l * DD * D2X, w1h,
                                     DD, D2X, KP1, NP1);
    wmma_gemm<<<g1, 128, 0, stream>>>(ah1, KP1, w1h, NP1,
                                      mlp_b1 + (size_t)l * D2X,
                                      tmp1, D2X, NN, D2X, KP1, 0);
    ln_relu<<<NN, 128, 0, stream>>>(tmp1, hdn, mlp_g + (size_t)l * D2X,
                                    mlp_beta + (size_t)l * D2X, D2X);
    cvt_a_k<<<bA2, 256, 0, stream>>>(hdn, ah2, NN, D2X, KP2);
    cvt_w_k<<<bW2, 256, 0, stream>>>(mlp_w2 + (size_t)l * D2X * DD, w2h,
                                     D2X, DD, KP2, NP2);
    wmma_gemm<<<g2, 128, 0, stream>>>(ah2, KP2, w2h, NP2,
                                      mlp_b2 + (size_t)l * DD,
                                      h, DD, NN, DD, KP2,
                                      (l == 0) ? 0 : 1);   // residual hc + hn
  }

  // final: relu(ln(h, blk_g[0], blk_b[0])) @ feat_w + feat_b
  ln_relu<<<NN, 128, 0, stream>>>(h, hn, blk_g, blk_b, DD);
  cvt_a_k<<<bA1, 256, 0, stream>>>(hn, ah1, NN, DD, KP1);
  cvt_w_k<<<bWF, 256, 0, stream>>>(feat_w, w1h, DD, FEATF, KP1, NPF);
  wmma_gemm<<<g3, 128, 0, stream>>>(ah1, KP1, w1h, NPF, feat_b,
                                    hfeat, FEATF, NN, FEATF, KP1, 0);

  // mean-pool per graph into d_out[0 : G*FEAT]
  float* pooled = (float*)d_out;
  zero_k<<<cdiv((long)GGN * FEATF, 256), 256, 0, stream>>>(pooled, (long)GGN * FEATF);
  zero_k<<<1, 256, 0, stream>>>(counts, GGN);
  pool_k<<<cdiv((long)NN * FEATF, 256), 256, 0, stream>>>(hfeat, batch, pooled, counts);
  pdiv_k<<<cdiv((long)GGN * FEATF, 256), 256, 0, stream>>>(pooled, counts);

  // prediction head into d_out[G*FEAT : G*FEAT + G*2]
  pred1_k<<<cdiv(GGN * FHN, 256), 256, 0, stream>>>(pooled, p1_w, p1_b, zbuf);
  pred2_k<<<1, 256, 0, stream>>>(zbuf, p2_w, p2_b, pooled + (size_t)GGN * FEATF);
}
